// Seq2SeqLSTM_25245817766586
// MI455X (gfx1250) — compile-verified, matrix operand order checked by reading, not run on hardware
//
#include <hip/hip_runtime.h>
#include <hip/hip_bf16.h>
#include <stdint.h>

// MI455X / gfx1250, wave32. bf16 WMMA 16x16x32 f32-acc; async global->LDS staging.
// Kernel shapes are compile-time specialized: (Kx=64,ldX=16384) enc-L0,
// (Kx=1024,ldX=1024) enc-L1/dec-L1, (Kx=0) dec-L0 (rank-1 scalar input).

typedef __bf16 bf16_t;
typedef __attribute__((ext_vector_type(16))) __bf16 v16bf;
typedef __attribute__((ext_vector_type(8)))  __bf16 bf16x8;
typedef __attribute__((ext_vector_type(8)))  float  v8f;

#define HID   1024
#define BATCH 256
#define SEQ   256
#define DIN   64
#define TOUT  32
#define KSLAB 512
#define LDSLD 520   // 512 + 8 bf16 padding per row

#if defined(__has_builtin)
#  if __has_builtin(__builtin_amdgcn_global_load_async_to_lds_b128) && \
      __has_builtin(__builtin_amdgcn_s_wait_asynccnt)
#    define USE_ASYNC_LDS 1
#  endif
#endif
#ifndef USE_ASYNC_LDS
#  define USE_ASYNC_LDS 0
#endif

#if USE_ASYNC_LDS
// builtin signature (probe-confirmed via hipcc diagnostic): pointers to
// int __attribute__((vector_size(16))) in global / LDS address spaces.
typedef int v4i __attribute__((vector_size(16)));
typedef __attribute__((address_space(1))) v4i* gv4i_p;
typedef __attribute__((address_space(3))) v4i* lv4i_p;
#endif

__device__ __forceinline__ bf16_t f2bf(float x) {
  union { float f; uint32_t u; } v; v.f = x;
  uint32_t r = v.u + 0x7FFFu + ((v.u >> 16) & 1u);   // round-to-nearest-even
  uint16_t h = (uint16_t)(r >> 16);
  return __builtin_bit_cast(bf16_t, h);
}

__device__ __forceinline__ float sigmoidf_(float x) {
  return 1.0f / (1.0f + __expf(-x));
}

// ---------------------------------------------------------------- f32 -> bf16
__global__ __launch_bounds__(256) void cvt_f32_to_bf16(
    const float* __restrict__ src, bf16_t* __restrict__ dst, int n) {
  int i = blockIdx.x * 256 + threadIdx.x;
  if (i < n) dst[i] = f2bf(src[i]);
}

// ------------------------------------------------------- one GEMM input phase
// A: 32 act rows staged into LDS in K-slabs via async DMA (ASYNCcnt-tracked),
//    shared by the 4 gate-waves; fragments read with ds_load_b128.
// B: W[n][k] row-major, per-lane column n = gate*H + jbase + (lane&15);
//    two 16B global loads per wmma, weights are L2-resident.
// 16-bit A/B fragment layout (ISA 7.12.2): elements 0..7 <- k = hi*8+0..7,
// elements 8..15 <- k = 16+hi*8+0..7 within each 32-wide K chunk.
template <int K, int LDA>
__device__ __forceinline__ void gemm_phase(
    const bf16_t* __restrict__ act, const bf16_t* __restrict__ W, int ldW,
    bf16_t* ldsA,
    int m0, int ncol, int tid, int lane, int hi, int mrow, v8f& acc) {
  constexpr int KW   = (K < KSLAB) ? K : KSLAB;  // slab width (compile-time)
  constexpr int CPR  = KW / 8;                   // 16B chunks per row
  constexpr int ITER = (32 * CPR) / 256;         // staging chunks per thread
  const int arow = mrow * 16 + (lane & 15);
  const bf16_t* wrow = W + (size_t)ncol * ldW;
#pragma unroll
  for (int ks = 0; ks < K; ks += KW) {
    __syncthreads();   // all waves done reading the previous slab
#pragma unroll
    for (int it = 0; it < ITER; ++it) {
      int idx  = tid + it * 256;
      int row  = idx / CPR;
      int col8 = (idx % CPR) * 8;
      const bf16_t* src = act + (size_t)(m0 + row) * LDA + (ks + col8);
      bf16_t* dst = ldsA + row * LDSLD + col8;
#if USE_ASYNC_LDS
      __builtin_amdgcn_global_load_async_to_lds_b128(
          (gv4i_p)const_cast<bf16_t*>(src), (lv4i_p)dst, 0, 0);
#else
      *reinterpret_cast<uint4*>(dst) = *reinterpret_cast<const uint4*>(src);
#endif
    }
#if USE_ASYNC_LDS
    __builtin_amdgcn_s_wait_asynccnt(0);
#endif
    __syncthreads();   // slab resident in LDS for every wave
    const bf16_t* wp0 = wrow + ks;
    const bf16_t* ap0 = ldsA + arow * LDSLD;
#pragma unroll 4
    for (int k0 = 0; k0 < KW; k0 += 32) {
      bf16x8 a0 = *reinterpret_cast<const bf16x8*>(ap0 + k0 + hi * 8);
      bf16x8 a1 = *reinterpret_cast<const bf16x8*>(ap0 + k0 + 16 + hi * 8);
      bf16x8 b0 = *reinterpret_cast<const bf16x8*>(wp0 + k0 + hi * 8);
      bf16x8 b1 = *reinterpret_cast<const bf16x8*>(wp0 + k0 + 16 + hi * 8);
      if (k0 + 32 < KW) __builtin_prefetch(wp0 + k0 + 32, 0, 3);  // near-scope
      v16bf a, b;
#pragma unroll
      for (int e = 0; e < 8; ++e) {
        a[e] = a0[e]; a[8 + e] = a1[e];
        b[e] = b0[e]; b[8 + e] = b1[e];
      }
      acc = __builtin_amdgcn_wmma_f32_16x16x32_bf16(
          false, a, false, b, (short)0, acc, false, false);
    }
  }
}

// -------------------------------------------- fused LSTM step (GEMM + cell)
// grid = (HID/16, BATCH/32), block = 256 (8 waves).
// Wave w: gate = w&3 (columns gate*H + jbase..+15), mrow = w>>2 (16 rows), so
// i/f/g/o for the same (b,j) land in this block's LDS for the fused cell.
template <int KX, int LDX>
__global__ __launch_bounds__(256) void lstm_step_kernel(
    const bf16_t* __restrict__ actX, const bf16_t* __restrict__ Wx,
    const bf16_t* __restrict__ actH, const bf16_t* __restrict__ Wh,
    const float* __restrict__ bias,
    const float* __restrict__ rU, const float* __restrict__ rV,
    float* __restrict__ cst, bf16_t* __restrict__ hout) {
  __shared__ __align__(16) bf16_t ldsA[32 * LDSLD];
  __shared__ float gLDS[4][32][16];

  const int tid  = threadIdx.x;
  const int lane = tid & 31;
  const int wid  = tid >> 5;
  const int gate = wid & 3;
  const int mrow = wid >> 2;
  const int hi   = lane >> 4;
  const int jbase = blockIdx.x * 16;
  const int m0    = blockIdx.y * 32;
  const int ncol  = gate * HID + jbase + (lane & 15);

  v8f acc;
#pragma unroll
  for (int i = 0; i < 8; ++i) acc[i] = 0.0f;

  if (KX > 0)
    gemm_phase<(KX > 0 ? KX : 32), LDX>(actX, Wx, KX, ldsA, m0, ncol,
                                        tid, lane, hi, mrow, acc);
  gemm_phase<HID, HID>(actH, Wh, HID, ldsA, m0, ncol, tid, lane, hi, mrow, acc);

  // C/D layout: lane = column n, VGPR v -> row hi*8+v
#pragma unroll
  for (int v = 0; v < 8; ++v)
    gLDS[gate][mrow * 16 + hi * 8 + v][lane & 15] = acc[v];
  __syncthreads();

#pragma unroll
  for (int e = 0; e < 2; ++e) {
    int idx = tid + e * 256;            // 512 elements, 256 threads
    int r = idx >> 4, col = idx & 15;
    int b = m0 + r, j = jbase + col;
    float gI = gLDS[0][r][col] + bias[j];
    float gF = gLDS[1][r][col] + bias[HID + j];
    float gG = gLDS[2][r][col] + bias[2 * HID + j];
    float gO = gLDS[3][r][col] + bias[3 * HID + j];
    if (KX == 0) {                       // decoder layer0 scalar-input rank-1 term
      float u = rU[b];
      gI += u * rV[j];
      gF += u * rV[HID + j];
      gG += u * rV[2 * HID + j];
      gO += u * rV[3 * HID + j];
    }
    size_t o = (size_t)b * HID + j;
    float cp = cst[o];
    float cn = sigmoidf_(gF) * cp + sigmoidf_(gI) * tanhf(gG);
    float hn = sigmoidf_(gO) * tanhf(cn);
    cst[o]  = cn;
    hout[o] = f2bf(hn);
  }
}

// -------------------------------------- pred = h1 @ fc_W^T + fc_b, feedback
__global__ __launch_bounds__(256) void pred_kernel(
    const bf16_t* __restrict__ h1, const float* __restrict__ fcW,
    const float* __restrict__ fcb, float* __restrict__ out,
    float* __restrict__ inp, int t) {
  __shared__ float red[256];
  int b = blockIdx.x, tid = threadIdx.x;
  const bf16_t* hr = h1 + (size_t)b * HID;
  float s = 0.0f;
  for (int k = tid; k < HID; k += 256) s += (float)hr[k] * fcW[k];
  red[tid] = s;
  __syncthreads();
  for (int o = 128; o > 0; o >>= 1) {
    if (tid < o) red[tid] += red[tid + o];
    __syncthreads();
  }
  if (tid == 0) {
    float p = red[0] + fcb[0];
    out[b * TOUT + t] = p;    // [B, T_OUT, 1]
    inp[b] = p;               // autoregressive feedback
  }
}

extern "C" void kernel_launch(void* const* d_in, const int* in_sizes, int n_in,
                              void* d_out, int out_size, void* d_ws, size_t ws_size,
                              hipStream_t stream) {
  (void)in_sizes; (void)n_in; (void)out_size; (void)ws_size;

  const float* X     = (const float*)d_in[0];
  const float* eWih0 = (const float*)d_in[1];
  const float* eWhh0 = (const float*)d_in[2];
  const float* eb0   = (const float*)d_in[3];
  const float* eWih1 = (const float*)d_in[4];
  const float* eWhh1 = (const float*)d_in[5];
  const float* eb1   = (const float*)d_in[6];
  const float* dWih0 = (const float*)d_in[7];   // [4H,1] -> rank-1 column, keep f32
  const float* dWhh0 = (const float*)d_in[8];
  const float* db0   = (const float*)d_in[9];
  const float* dWih1 = (const float*)d_in[10];
  const float* dWhh1 = (const float*)d_in[11];
  const float* db1   = (const float*)d_in[12];
  const float* fcW   = (const float*)d_in[13];
  const float* fcb   = (const float*)d_in[14];
  float* out = (float*)d_out;

  char* ws = (char*)d_ws;
  size_t off = 0;
  auto alloc = [&](size_t bytes) -> char* {
    char* p = ws + off;
    off = (off + bytes + 255) & ~(size_t)255;
    return p;
  };

  const size_t WBIG = (size_t)4 * HID * HID;      // 4M elements
  bf16_t* Xbf    = (bf16_t*)alloc((size_t)BATCH * SEQ * DIN * 2);
  bf16_t* eWih0b = (bf16_t*)alloc((size_t)4 * HID * DIN * 2);
  bf16_t* eWhh0b = (bf16_t*)alloc(WBIG * 2);
  bf16_t* eWih1b = (bf16_t*)alloc(WBIG * 2);
  bf16_t* eWhh1b = (bf16_t*)alloc(WBIG * 2);
  bf16_t* dWhh0b = (bf16_t*)alloc(WBIG * 2);
  bf16_t* dWih1b = (bf16_t*)alloc(WBIG * 2);
  bf16_t* dWhh1b = (bf16_t*)alloc(WBIG * 2);
  bf16_t* h0a = (bf16_t*)alloc((size_t)BATCH * HID * 2);
  bf16_t* h0b = (bf16_t*)alloc((size_t)BATCH * HID * 2);
  bf16_t* h1a = (bf16_t*)alloc((size_t)BATCH * HID * 2);
  bf16_t* h1b = (bf16_t*)alloc((size_t)BATCH * HID * 2);
  float*  c0  = (float*)alloc((size_t)BATCH * HID * 4);
  float*  c1  = (float*)alloc((size_t)BATCH * HID * 4);
  float*  inp = (float*)alloc((size_t)BATCH * 4);

  auto cvt = [&](const float* s, bf16_t* d, int n) {
    cvt_f32_to_bf16<<<(n + 255) / 256, 256, 0, stream>>>(s, d, n);
  };
  cvt(X,     Xbf,    BATCH * SEQ * DIN);
  cvt(eWih0, eWih0b, 4 * HID * DIN);
  cvt(eWhh0, eWhh0b, (int)WBIG);
  cvt(eWih1, eWih1b, (int)WBIG);
  cvt(eWhh1, eWhh1b, (int)WBIG);
  cvt(dWhh0, dWhh0b, (int)WBIG);
  cvt(dWih1, dWih1b, (int)WBIG);
  cvt(dWhh1, dWhh1b, (int)WBIG);

  (void)hipMemsetAsync(h0a, 0, (size_t)BATCH * HID * 2, stream);
  (void)hipMemsetAsync(h1a, 0, (size_t)BATCH * HID * 2, stream);
  (void)hipMemsetAsync(c0,  0, (size_t)BATCH * HID * 4, stream);
  (void)hipMemsetAsync(c1,  0, (size_t)BATCH * HID * 4, stream);
  (void)hipMemsetAsync(inp, 0, (size_t)BATCH * 4, stream);

  dim3 grid(HID / 16, BATCH / 32);   // 64 x 8 blocks, 256 threads

  bf16_t* h0c = h0a; bf16_t* h0n = h0b;
  bf16_t* h1c = h1a; bf16_t* h1n = h1b;

  // ---------------- encoder: 2 stacked layers, streamed per time step
  for (int t = 0; t < SEQ; ++t) {
    lstm_step_kernel<DIN, SEQ * DIN><<<grid, 256, 0, stream>>>(
        Xbf + (size_t)t * DIN, eWih0b,
        h0c, eWhh0b, eb0, nullptr, nullptr, c0, h0n);
    { bf16_t* tmp = h0c; h0c = h0n; h0n = tmp; }
    lstm_step_kernel<HID, HID><<<grid, 256, 0, stream>>>(
        h0c, eWih1b,
        h1c, eWhh1b, eb1, nullptr, nullptr, c1, h1n);
    { bf16_t* tmp = h1c; h1c = h1n; h1n = tmp; }
  }

  // ---------------- autoregressive decoder
  for (int t = 0; t < TOUT; ++t) {
    lstm_step_kernel<0, HID><<<grid, 256, 0, stream>>>(
        nullptr, nullptr,
        h0c, dWhh0b, db0, inp, dWih0, c0, h0n);
    { bf16_t* tmp = h0c; h0c = h0n; h0n = tmp; }
    lstm_step_kernel<HID, HID><<<grid, 256, 0, stream>>>(
        h0c, dWih1b,
        h1c, dWhh1b, db1, nullptr, nullptr, c1, h1n);
    { bf16_t* tmp = h1c; h1c = h1n; h1n = tmp; }
    pred_kernel<<<BATCH, 256, 0, stream>>>(h1c, fcW, fcb, out, inp, t);
  }
}
